// MyLSTM_19748259627793
// MI455X (gfx1250) — matrix operand order repaired and checked by reading, not compiled, not run on hardware
//
#include <hip/hip_runtime.h>

typedef __attribute__((ext_vector_type(16))) __bf16 v16bf;
typedef __attribute__((ext_vector_type(8)))  float  v8f;

struct __align__(16) U4 { unsigned int x, y, z, w; };
union Frag16 { v16bf v; U4 q[2]; };

constexpr int T_STEPS = 256, B_SZ = 512, IN_SZ = 8, H_SZ = 256;
constexpr int ROWS = 16;                 // batch rows per block
constexpr int FRAG_ELEMS = 512;          // 32 lanes * 16 bf16 per fragment
constexpr int WS_V_FRAGS = 4 * 16 * 8;   // gate * ntile * kstep
constexpr int WS_W_FRAGS = 4 * 16;       // gate * ntile (K padded to 32)

// ---------------------------------------------------------------------------
// Prep: convert V_g (H x H f32) and W_g (H x 8 f32) into bf16 B-fragments in
// per-lane-contiguous order. B 32x16 layout: lane = half*16 + n, k = 16*half+e.
// Strides (region V): e=1, lane=16, ks=512, nt=4096, g=65536.
// ---------------------------------------------------------------------------
__global__ __launch_bounds__(256) void lstm_prep(
    const float* __restrict__ Vf, const float* __restrict__ Vi,
    const float* __restrict__ Vc, const float* __restrict__ Vo,
    const float* __restrict__ Wf, const float* __restrict__ Wi,
    const float* __restrict__ Wc, const float* __restrict__ Wo,
    __bf16* __restrict__ ws)
{
  int idx = blockIdx.x * 256 + threadIdx.x;
  const int totalV = WS_V_FRAGS * FRAG_ELEMS;   // 262144 = 2^18
  const int totalW = WS_W_FRAGS * FRAG_ELEMS;   // 32768  = 2^15
  if (idx < totalV) {
    int e    = idx & 15;
    int lane = (idx >> 4) & 31;
    int ks   = (idx >> 9) & 7;
    int nt   = (idx >> 12) & 15;
    int g    = idx >> 16;
    int n    = nt * 16 + (lane & 15);
    int half = lane >> 4;
    int k    = ks * 32 + 16 * half + e;
    const float* V = (g == 0) ? Vf : (g == 1) ? Vi : (g == 2) ? Vc : Vo;
    ws[idx] = (__bf16)V[n * H_SZ + k];
  } else if (idx < totalV + totalW) {
    int j    = idx - totalV;
    int e    = j & 15;
    int lane = (j >> 4) & 31;
    int nt   = (j >> 9) & 15;
    int g    = j >> 13;
    int n    = nt * 16 + (lane & 15);
    int half = lane >> 4;
    float val = 0.0f;
    if (half == 0 && e < 8) {             // k = e < IN_SZ, rest zero-padded
      const float* W = (g == 0) ? Wf : (g == 1) ? Wi : (g == 2) ? Wc : Wo;
      val = W[n * IN_SZ + e];
    }
    ws[idx] = (__bf16)val;
  }
}

__device__ __forceinline__ float sigm(float x) { return 1.0f / (1.0f + __expf(-x)); }

// ---------------------------------------------------------------------------
// Persistent recurrence kernel: 32 blocks x 16 batch rows; h (bf16) lives in
// LDS, c lives in VGPRs in WMMA C-fragment layout, for all 256 timesteps.
// One N-tile at a time; an opaque per-iteration offset keeps the weight
// fragment loads inside the loop (L2-resident GLOBAL stream, LOADcnt only)
// without breaking address-space inference.
// ---------------------------------------------------------------------------
__global__ __launch_bounds__(256, 1) void lstm_main(
    const float* __restrict__ x,  const float* __restrict__ h0, const float* __restrict__ c0,
    const float* __restrict__ bf_, const float* __restrict__ bi_,
    const float* __restrict__ bc_, const float* __restrict__ bo_,
    const float* __restrict__ mf_, const float* __restrict__ mi_,
    const float* __restrict__ mc_, const float* __restrict__ mo_,
    const float* __restrict__ Wout, const float* __restrict__ bout,
    const __bf16* __restrict__ wsV, const __bf16* __restrict__ wsW,
    float* __restrict__ out)
{
  __shared__ __align__(16) __bf16 hS[ROWS * H_SZ];     // 8 KB
  __shared__ __align__(16) __bf16 xS[ROWS * IN_SZ];    // 256 B
  __shared__ __bf16 maskS[4][ROWS * H_SZ];             // 32 KB
  __shared__ float  biasS[4][H_SZ];                    // 4 KB
  __shared__ float  woutS[H_SZ];                       // 1 KB
  __shared__ float  partS[ROWS][16];                   // 1 KB

  const int tid  = threadIdx.x;
  const int rb   = blockIdx.x * ROWS;
  const int wave = tid >> 5;
  const int lane = tid & 31;
  const int half = lane >> 4;
  const int nlo  = lane & 15;

  const float* masks[4]  = { mf_, mi_, mc_, mo_ };
  const float* biases[4] = { bf_, bi_, bc_, bo_ };
  for (int g = 0; g < 4; ++g) {
    for (int i = tid; i < ROWS * H_SZ; i += 256) {
      int m = i >> 8, n = i & 255;
      maskS[g][i] = (__bf16)masks[g][(rb + m) * H_SZ + n];
    }
    for (int i = tid; i < H_SZ; i += 256) biasS[g][i] = biases[g][i];
  }
  for (int i = tid; i < H_SZ; i += 256) woutS[i] = Wout[i];
  for (int i = tid; i < ROWS * H_SZ; i += 256) {
    int m = i >> 8, n = i & 255;
    hS[i] = (__bf16)h0[(rb + m) * H_SZ + n];
  }
  // c state in C/D fragment layout: lane(n) = nlo of n-tile, row m = r + 8*half
  float creg[2][8];
  #pragma unroll
  for (int j = 0; j < 2; ++j) {
    int ng = (wave * 2 + j) * 16 + nlo;
    #pragma unroll
    for (int r = 0; r < 8; ++r)
      creg[j][r] = c0[(rb + r + 8 * half) * H_SZ + ng];
  }
  const float bo_s = bout[0];
  __syncthreads();

  // opaque zero offset: asm-redefined per iteration so weight-fragment loads
  // are not loop-invariant (no LICM -> no scratch spill), while the base
  // pointers keep their kernarg->global provenance (global_load, LOADcnt only)
  unsigned long long opq = 0;

  for (int t = 0; t < T_STEPS; ++t) {
    asm volatile("" : "+s"(opq));
    const __bf16* pV = wsV + opq;
    const __bf16* pW = wsW + opq;

    // stage x_t tile (16 x 8) as bf16
    if (tid < ROWS * IN_SZ) {
      int m = tid >> 3, k = tid & 7;
      xS[tid] = (__bf16)x[((t * B_SZ) + rb + m) * IN_SZ + k];
    }
    __syncthreads();

    // A fragment for padded x@W^T (K=32, only k<8 nonzero -> half==0, e<8)
    Frag16 ax;
    ax.q[1] = U4{0, 0, 0, 0};
    if (half == 0) ax.q[0] = *(const U4*)&xS[nlo * IN_SZ];
    else           ax.q[0] = U4{0, 0, 0, 0};

    // A fragments for h (16x32 bf16 per kstep): two contiguous 16B LDS loads
    Frag16 ah[8];
    #pragma unroll
    for (int ks = 0; ks < 8; ++ks) {
      int base = nlo * H_SZ + ks * 32 + 16 * half;
      ah[ks].q[0] = *(const U4*)&hS[base];
      ah[ks].q[1] = *(const U4*)&hS[base + 8];
    }
    __syncthreads();   // all hS/xS reads issued+consumed before any hS write

    #pragma unroll
    for (int j = 0; j < 2; ++j) {
      const int nt = wave * 2 + j;
      const int ng = nt * 16 + nlo;

      v8f acc[4];
      #pragma unroll
      for (int g = 0; g < 4; ++g) {
        v8f a = {};   // SRC2 = inline 0, no register splat
        v16bf bw = *(const v16bf*)(pW + ((g * 16 + nt) * FRAG_ELEMS + lane * 16));
        a = __builtin_amdgcn_wmma_f32_16x16x32_bf16(false, ax.v, false, bw,
                                                    (short)0, a, false, false);
        #pragma unroll
        for (int ks = 0; ks < 8; ++ks) {
          v16bf bvv = *(const v16bf*)(pV + (((g * 16 + nt) * 8 + ks) * FRAG_ELEMS + lane * 16));
          a = __builtin_amdgcn_wmma_f32_16x16x32_bf16(false, ah[ks].v, false, bvv,
                                                      (short)0, a, false, false);
        }
        acc[g] = a;
      }

      // bias + sigmoid + static dropout + cell/hidden update for this n-tile
      float bv0 = biasS[0][ng], bv1 = biasS[1][ng];
      float bv2 = biasS[2][ng], bv3 = biasS[3][ng];
      #pragma unroll
      for (int r = 0; r < 8; ++r) {
        int m  = r + 8 * half;
        int mn = m * H_SZ + ng;
        float fg = sigm(acc[0][r] + bv0) * (float)maskS[0][mn];
        float ig = sigm(acc[1][r] + bv1) * (float)maskS[1][mn];
        float gg = sigm(acc[2][r] + bv2) * (float)maskS[2][mn];
        float og = sigm(acc[3][r] + bv3) * (float)maskS[3][mn];
        float cn = fg * creg[j][r] + ig * gg;
        creg[j][r] = cn;
        hS[mn] = (__bf16)(og * tanhf(cn));
      }
    }
    __syncthreads();

    // returns[t] = tanh(h @ W_out^T + b_out): 16 threads/row partial reduce
    {
      int row = tid >> 4, seg = tid & 15;
      float s = 0.0f;
      #pragma unroll
      for (int k = 0; k < 16; ++k)
        s += (float)hS[row * H_SZ + seg * 16 + k] * woutS[seg * 16 + k];
      partS[row][seg] = s;
    }
    __syncthreads();
    if (tid < ROWS) {
      float s = bo_s;
      #pragma unroll
      for (int k = 0; k < 16; ++k) s += partS[tid][k];
      out[t * B_SZ + rb + tid] = tanhf(s);
    }
    __syncthreads();
  }
}

extern "C" void kernel_launch(void* const* d_in, const int* in_sizes, int n_in,
                              void* d_out, int out_size, void* d_ws, size_t ws_size,
                              hipStream_t stream) {
  const float* x    = (const float*)d_in[0];
  const float* h0   = (const float*)d_in[1];
  const float* c0   = (const float*)d_in[2];
  const float* W_f  = (const float*)d_in[3];
  const float* V_f  = (const float*)d_in[4];
  const float* W_i  = (const float*)d_in[5];
  const float* V_i  = (const float*)d_in[6];
  const float* W_c  = (const float*)d_in[7];
  const float* V_c  = (const float*)d_in[8];
  const float* W_o  = (const float*)d_in[9];
  const float* V_o  = (const float*)d_in[10];
  const float* b_f  = (const float*)d_in[11];
  const float* b_i  = (const float*)d_in[12];
  const float* b_c  = (const float*)d_in[13];
  const float* b_o  = (const float*)d_in[14];
  const float* m_f  = (const float*)d_in[15];
  const float* m_i  = (const float*)d_in[16];
  const float* m_c  = (const float*)d_in[17];
  const float* m_o  = (const float*)d_in[18];
  const float* Wout = (const float*)d_in[19];
  const float* bout = (const float*)d_in[20];

  __bf16* ws = (__bf16*)d_ws;
  const __bf16* wsV = ws;
  const __bf16* wsW = ws + (size_t)WS_V_FRAGS * FRAG_ELEMS;

  int totalPrep = (WS_V_FRAGS + WS_W_FRAGS) * FRAG_ELEMS;   // 294912 bf16
  lstm_prep<<<(totalPrep + 255) / 256, 256, 0, stream>>>(
      V_f, V_i, V_c, V_o, W_f, W_i, W_c, W_o, ws);

  lstm_main<<<B_SZ / ROWS, 256, 0, stream>>>(
      x, h0, c0, b_f, b_i, b_c, b_o, m_f, m_i, m_c, m_o,
      Wout, bout, wsV, wsW, (float*)d_out);
}
